// BlipAttention_86938728005695
// MI455X (gfx1250) — compile-verified, matrix operand order checked by reading, not run on hardware
//
#include <hip/hip_runtime.h>
#include <hip/hip_bf16.h>

// ---------------------------------------------------------------------------
// BlipAttention for MI455X (gfx1250): bf16 WMMA, f32 accumulate.
//   B=32, N=1024, C=768, 3C=2304.
// Round 3: add Tensor Data Mover (tensor_load_to_lds + s_wait_tensorcnt) for
// the attention Q-tile staging (per-row descriptors preserve LDS padding),
// with a cooperative-copy fallback if the builtin is unavailable.
// ---------------------------------------------------------------------------

typedef __bf16 bf16_t;
typedef __attribute__((ext_vector_type(16))) __bf16 v16bf;
typedef __attribute__((ext_vector_type(8)))  __bf16 v8bf;
typedef __attribute__((ext_vector_type(8)))  float  v8f;
typedef __attribute__((ext_vector_type(4)))  unsigned int v4u;
typedef __attribute__((ext_vector_type(8)))  int v8i;
typedef __attribute__((ext_vector_type(4)))  int v4i;

#define DEV __device__ __forceinline__

#if defined(__has_builtin)
#if __has_builtin(__builtin_amdgcn_tensor_load_to_lds) && \
    __has_builtin(__builtin_amdgcn_s_wait_tensorcnt)
#define HAVE_TDM 1
#endif
#endif
#ifndef HAVE_TDM
#define HAVE_TDM 0
#endif

static constexpr int BATCH = 32;
static constexpr int SEQ   = 1024;
static constexpr int C     = 768;
static constexpr int C3    = 2304;
static constexpr int MTOT  = BATCH * SEQ;     // 32768
static constexpr int KCH   = C / 32;          // 24 k-chunks per 768 dot

// Padded LDS strides (64 banks x 4B). Row offsets chosen so 16-lane b128
// fragment reads / 16-thread softmax rows stripe all banks.
static constexpr int SLD = SEQ + 16;          // f32 score row stride (≡16 banks)
static constexpr int PLD = SEQ + 8;           // bf16 prob row stride (≡4 words)
static constexpr int QLD = C + 8;             // bf16 Q row stride  (≡4 words)

// Load one 16x32 bf16 WMMA fragment (A, or B given B^T row-major storage).
// ISA 16-bit layout: lane L (L%16 = row), K runs {h*8..+7, 16+h*8..+7}, h=L/16.
DEV v16bf load_frag(const bf16_t* M, int ld, int row0, int k0) {
    const int lane = threadIdx.x & 31;
    const bf16_t* p = M + (long)(row0 + (lane & 15)) * ld + k0 + ((lane >> 4) << 3);
    v8bf lo = *(const v8bf*)p;          // K = h*8 .. h*8+7
    v8bf hi = *(const v8bf*)(p + 16);   // K = 16+h*8 .. +7
    return __builtin_shufflevector(lo, hi, 0,1,2,3,4,5,6,7,8,9,10,11,12,13,14,15);
}

DEV v8f wmma_bf16(v16bf a, v16bf b, v8f c) {
    return __builtin_amdgcn_wmma_f32_16x16x32_bf16(
        false, a, false, b, (short)0, c, false, false);
}

#if HAVE_TDM
// TDM: DMA one row of `nelem` bf16 from global into LDS at byte addr lds_addr.
// D# per CDNA5 ISA ch8: group0 {count=1, lds_addr, gaddr lo, gaddr[56:32]|type=2},
// group1 {data_size=2B, tensor_dim0=nelem, tensor_dim1=1, tile_dim0=nelem,
//         tile_dim1=1, dim0_stride=nelem}; groups 2/3 zero (dims <= 2D).
DEV void tdm_load_row(const void* gsrc, unsigned lds_addr, unsigned nelem) {
    unsigned long long ga = (unsigned long long)(uintptr_t)gsrc;
    v4u g0 = { 1u,                                   // count=1 (valid user D#)
               lds_addr,
               (unsigned)ga,
               (unsigned)((ga >> 32) & 0x1FFFFFFu) | 0x80000000u };  // type=2
    v8i g1 = { (int)0x00010000,                      // data_size=1 (2 bytes)
               (int)((nelem & 0xFFFFu) << 16),       // tensor_dim0[15:0]
               (int)(1u << 16),                      // tensor_dim1 = 1
               (int)((nelem & 0xFFFFu) << 16),       // tile_dim0
               (int)1,                               // tile_dim1 = 1
               (int)nelem,                           // tensor_dim0_stride[31:0]
               0, 0 };
    v4i g2 = { 0, 0, 0, 0 };
    v4i g3 = { 0, 0, 0, 0 };
#if __clang_major__ >= 23
    v8i g4 = { 0, 0, 0, 0, 0, 0, 0, 0 };
    __builtin_amdgcn_tensor_load_to_lds(g0, g1, g2, g3, g4, 0);
#else
    __builtin_amdgcn_tensor_load_to_lds(g0, g1, g2, g3, 0);
#endif
}
#endif

// ---------------------------------------------------------------------------
// Conversion kernels
// ---------------------------------------------------------------------------
__global__ void k_conv_bf16(const float* __restrict__ in, bf16_t* __restrict__ out, long n) {
    long i = (long)blockIdx.x * blockDim.x + threadIdx.x;
    if (i < n) out[i] = (bf16_t)in[i];
}

// in: [rows][cols] f32  ->  out: [cols][rows] bf16
__global__ void k_conv_bf16_t(const float* __restrict__ in, bf16_t* __restrict__ out,
                              int rows, int cols) {
    long i = (long)blockIdx.x * blockDim.x + threadIdx.x;
    long n = (long)rows * cols;
    if (i < n) {
        int r = (int)(i / cols), c = (int)(i % cols);
        out[(long)c * rows + r] = (bf16_t)in[i];
    }
}

// ---------------------------------------------------------------------------
// QKV GEMM: 64x32 output per wave (4x2 tiles), 8 waves/block.
// ---------------------------------------------------------------------------
__global__ __launch_bounds__(256) void k_qkv_gemm(
    const bf16_t* __restrict__ xb, const bf16_t* __restrict__ wt,
    const float* __restrict__ bias,
    bf16_t* __restrict__ qb, bf16_t* __restrict__ kb, bf16_t* __restrict__ vtb) {
    const int wave = threadIdx.x >> 5;
    const int lane = threadIdx.x & 31;
    const long wtile = (long)blockIdx.x * 8 + wave;   // 512 * 72 wave-tiles
    const int m0 = (int)(wtile / 72) * 64;
    const int n0 = (int)(wtile % 72) * 32;

    v8f acc[4][2] = {};
    for (int kc = 0; kc < KCH; ++kc) {
        const int k0 = kc * 32;
        v16bf a0 = load_frag(xb, C, m0,      k0);
        v16bf a1 = load_frag(xb, C, m0 + 16, k0);
        v16bf a2 = load_frag(xb, C, m0 + 32, k0);
        v16bf a3 = load_frag(xb, C, m0 + 48, k0);
        v16bf b0 = load_frag(wt, C, n0,      k0);
        v16bf b1 = load_frag(wt, C, n0 + 16, k0);
        acc[0][0] = wmma_bf16(a0, b0, acc[0][0]);
        acc[0][1] = wmma_bf16(a0, b1, acc[0][1]);
        acc[1][0] = wmma_bf16(a1, b0, acc[1][0]);
        acc[1][1] = wmma_bf16(a1, b1, acc[1][1]);
        acc[2][0] = wmma_bf16(a2, b0, acc[2][0]);
        acc[2][1] = wmma_bf16(a2, b1, acc[2][1]);
        acc[3][0] = wmma_bf16(a3, b0, acc[3][0]);
        acc[3][1] = wmma_bf16(a3, b1, acc[3][1]);
    }

    const int Nc = lane & 15;
    const int hm = (lane >> 4) << 3;
    const float qscale = 0.03608439182435161f;   // 1/sqrt(768)
    #pragma unroll
    for (int j = 0; j < 2; ++j) {
        const int n  = n0 + j * 16 + Nc;
        const float bv = bias[n];
        #pragma unroll
        for (int i = 0; i < 4; ++i) {
            #pragma unroll
            for (int e = 0; e < 8; ++e) {
                const int m = m0 + i * 16 + e + hm;
                const float v = acc[i][j][e] + bv;
                if (n < C) {
                    qb[(long)m * C + n] = (bf16_t)(v * qscale);
                } else if (n < 2 * C) {
                    kb[(long)m * C + (n - C)] = (bf16_t)v;
                } else {
                    const int bi = m >> 10, tok = m & 1023, c = n - 2 * C;
                    vtb[((long)bi * C + c) * SEQ + tok] = (bf16_t)v;  // V^T [b][c][n]
                }
            }
        }
    }
}

// ---------------------------------------------------------------------------
// Attention: one block = (batch, 16-query tile). 8 waves, 256 threads.
// ---------------------------------------------------------------------------
__global__ __launch_bounds__(256) void k_attn(
    const bf16_t* __restrict__ qb, const bf16_t* __restrict__ kb,
    const bf16_t* __restrict__ vtb, bf16_t* __restrict__ ab) {
    __shared__ float  S[16 * SLD];     // ~65 KB score tile (padded rows)
    __shared__ bf16_t P[16 * PLD];     // ~32 KB probability tile (padded)
    __shared__ bf16_t Qs[16 * QLD];    // ~24 KB staged Q tile (padded)
    __shared__ float  red[16 * 16];
    __shared__ float  rstat[16];

    const int b    = blockIdx.x >> 6;
    const int q0   = (blockIdx.x & 63) << 4;
    const int wave = threadIdx.x >> 5;
    const int lane = threadIdx.x & 31;
    const int qrow = b * SEQ + q0;

    // ---- Stage Q tile (16 x 768 bf16) into LDS
#if HAVE_TDM
    {   // TDM DMA: 2 rows per wave, padded LDS row stride kept via per-row D#
        const int r0 = wave * 2;
        const bf16_t* src = qb + (long)(qrow + r0) * C;
        const unsigned lbase = (unsigned)(uintptr_t)(void*)&Qs[0];
        tdm_load_row(src,     lbase + (unsigned)(r0    ) * (QLD * 2), C);
        tdm_load_row(src + C, lbase + (unsigned)(r0 + 1) * (QLD * 2), C);
        __builtin_amdgcn_s_wait_tensorcnt(0);
    }
#else
    {
        const bf16_t* src = qb + (long)qrow * C;
        for (int idx = threadIdx.x; idx < 16 * (C / 8); idx += 256) {
            const int r = idx / (C / 8), cc = (idx % (C / 8)) * 8;
            *(v8bf*)&Qs[r * QLD + cc] = *(const v8bf*)(src + (long)r * C + cc);
        }
    }
#endif
    __syncthreads();

    // ---- Phase 1: S = (Q*scale).K^T ; each wave: 8 key tiles, 2 at a time
    #pragma unroll
    for (int u = 0; u < 4; ++u) {
        const int na = (wave * 8 + u * 2) * 16;
        const int nb = na + 16;
        v8f acc0 = {}, acc1 = {};
        for (int kc = 0; kc < KCH; ++kc) {
            const int k0 = kc * 32;
            v16bf a  = load_frag(Qs, QLD, 0, k0);                // from LDS
            v16bf b0 = load_frag(kb, C, b * SEQ + na, k0);
            v16bf b1 = load_frag(kb, C, b * SEQ + nb, k0);
            acc0 = wmma_bf16(a, b0, acc0);
            acc1 = wmma_bf16(a, b1, acc1);
        }
        const int Nc = lane & 15, hm = (lane >> 4) << 3;
        #pragma unroll
        for (int e = 0; e < 8; ++e) {
            const int M = e + hm;
            S[M * SLD + na + Nc] = acc0[e];
            S[M * SLD + nb + Nc] = acc1[e];
        }
    }
    __syncthreads();

    // ---- Phase 2: exact row softmax (16 threads per row, 64 elems each)
    const int r   = threadIdx.x >> 4;
    const int c16 = threadIdx.x & 15;
    float* row = &S[r * SLD];

    float mx = -3.0e38f;
    for (int j = c16 * 64; j < c16 * 64 + 64; ++j) mx = fmaxf(mx, row[j]);
    red[r * 16 + c16] = mx;
    __syncthreads();
    if (c16 == 0) {
        float m2 = red[r * 16];
        for (int t = 1; t < 16; ++t) m2 = fmaxf(m2, red[r * 16 + t]);
        rstat[r] = m2;
    }
    __syncthreads();
    const float rowmax = rstat[r];
    float s = 0.f;
    for (int j = c16 * 64; j < c16 * 64 + 64; ++j) {
        float e = __expf(row[j] - rowmax);
        row[j] = e;
        s += e;
    }
    red[r * 16 + c16] = s;
    __syncthreads();
    if (c16 == 0) {
        float t2 = 0.f;
        for (int t = 0; t < 16; ++t) t2 += red[r * 16 + t];
        rstat[r] = 1.0f / t2;
    }
    __syncthreads();
    const float inv = rstat[r];
    for (int j = c16 * 64; j < c16 * 64 + 64; ++j)
        P[r * PLD + j] = (bf16_t)(row[j] * inv);
    __syncthreads();

    // ---- Phase 3: O = P.V ; each wave: 6 channel tiles, 2 at a time
    #pragma unroll
    for (int u = 0; u < 3; ++u) {
        const int c0 = (wave * 6 + u * 2) * 16;
        const int c1 = c0 + 16;
        v8f acc0 = {}, acc1 = {};
        for (int kk = 0; kk < SEQ / 32; ++kk) {
            const int k0 = kk * 32;
            v16bf a  = load_frag(P, PLD, 0, k0);                 // from LDS
            v16bf b0 = load_frag(vtb, SEQ, b * C + c0, k0);
            v16bf b1 = load_frag(vtb, SEQ, b * C + c1, k0);
            acc0 = wmma_bf16(a, b0, acc0);
            acc1 = wmma_bf16(a, b1, acc1);
        }
        const int Nc = lane & 15, hm = (lane >> 4) << 3;
        #pragma unroll
        for (int e = 0; e < 8; ++e) {
            const int M = e + hm;
            ab[(long)(qrow + M) * C + c0 + Nc] = (bf16_t)acc0[e];
            ab[(long)(qrow + M) * C + c1 + Nc] = (bf16_t)acc1[e];
        }
    }
}

// ---------------------------------------------------------------------------
// Projection GEMM: 64x32 per wave; out f32 = ab @ WpT + bias
// ---------------------------------------------------------------------------
__global__ __launch_bounds__(256) void k_proj_gemm(
    const bf16_t* __restrict__ ab, const bf16_t* __restrict__ wt,
    const float* __restrict__ bias, float* __restrict__ out) {
    const int wave = threadIdx.x >> 5;
    const int lane = threadIdx.x & 31;
    const long wtile = (long)blockIdx.x * 8 + wave;   // 512 * 24 wave-tiles
    const int m0 = (int)(wtile / 24) * 64;
    const int n0 = (int)(wtile % 24) * 32;

    v8f acc[4][2] = {};
    for (int kc = 0; kc < KCH; ++kc) {
        const int k0 = kc * 32;
        v16bf a0 = load_frag(ab, C, m0,      k0);
        v16bf a1 = load_frag(ab, C, m0 + 16, k0);
        v16bf a2 = load_frag(ab, C, m0 + 32, k0);
        v16bf a3 = load_frag(ab, C, m0 + 48, k0);
        v16bf b0 = load_frag(wt, C, n0,      k0);
        v16bf b1 = load_frag(wt, C, n0 + 16, k0);
        acc[0][0] = wmma_bf16(a0, b0, acc[0][0]);
        acc[0][1] = wmma_bf16(a0, b1, acc[0][1]);
        acc[1][0] = wmma_bf16(a1, b0, acc[1][0]);
        acc[1][1] = wmma_bf16(a1, b1, acc[1][1]);
        acc[2][0] = wmma_bf16(a2, b0, acc[2][0]);
        acc[2][1] = wmma_bf16(a2, b1, acc[2][1]);
        acc[3][0] = wmma_bf16(a3, b0, acc[3][0]);
        acc[3][1] = wmma_bf16(a3, b1, acc[3][1]);
    }

    const int Nc = lane & 15, hm = (lane >> 4) << 3;
    #pragma unroll
    for (int j = 0; j < 2; ++j) {
        const int n  = n0 + j * 16 + Nc;
        const float bv = bias[n];
        #pragma unroll
        for (int i = 0; i < 4; ++i) {
            #pragma unroll
            for (int e = 0; e < 8; ++e) {
                const int m = m0 + i * 16 + e + hm;
                out[(long)m * C + n] = acc[i][j][e] + bv;
            }
        }
    }
}

// ---------------------------------------------------------------------------
extern "C" void kernel_launch(void* const* d_in, const int* in_sizes, int n_in,
                              void* d_out, int out_size, void* d_ws, size_t ws_size,
                              hipStream_t stream) {
    const float* x      = (const float*)d_in[0];
    const float* W_qkv  = (const float*)d_in[1];
    const float* b_qkv  = (const float*)d_in[2];
    const float* W_proj = (const float*)d_in[3];
    const float* b_proj = (const float*)d_in[4];
    float* out = (float*)d_out;

    size_t off = 0;
    auto carve = [&](size_t bytes) -> char* {
        char* p = (char*)d_ws + off;
        off = (off + bytes + 255) & ~(size_t)255;
        return p;
    };
    bf16_t* xb    = (bf16_t*)carve((size_t)MTOT * C * 2);
    bf16_t* wqkvt = (bf16_t*)carve((size_t)C3 * C * 2);
    bf16_t* wprjt = (bf16_t*)carve((size_t)C * C * 2);
    bf16_t* qbuf  = (bf16_t*)carve((size_t)MTOT * C * 2);
    bf16_t* kbuf  = (bf16_t*)carve((size_t)MTOT * C * 2);
    bf16_t* vtb   = (bf16_t*)carve((size_t)MTOT * C * 2);
    bf16_t* abuf  = (bf16_t*)carve((size_t)MTOT * C * 2);
    (void)ws_size; (void)in_sizes; (void)n_in; (void)out_size;

    {   // conversions
        long n = (long)MTOT * C;
        k_conv_bf16<<<(int)((n + 255) / 256), 256, 0, stream>>>(x, xb, n);
        long nq = (long)C * C3;
        k_conv_bf16_t<<<(int)((nq + 255) / 256), 256, 0, stream>>>(W_qkv, wqkvt, C, C3);
        long np = (long)C * C;
        k_conv_bf16_t<<<(int)((np + 255) / 256), 256, 0, stream>>>(W_proj, wprjt, C, C);
    }

    // QKV GEMM: (32768/64)*(2304/32) = 512*72 wave-tiles, 8 per block
    k_qkv_gemm<<<(512 * 72) / 8, 256, 0, stream>>>(xb, wqkvt, b_qkv, qbuf, kbuf, vtb);

    // attention: 32 batches * 64 query tiles
    k_attn<<<BATCH * (SEQ / 16), 256, 0, stream>>>(qbuf, kbuf, vtb, abuf);

    // projection GEMM: (32768/64)*(768/32) = 512*24 wave-tiles, 8 per block
    k_proj_gemm<<<(512 * 24) / 8, 256, 0, stream>>>(abuf, wprjt, b_proj, out);
}
